// YOLOLoss_8022998909594
// MI455X (gfx1250) — compile-verified, matrix operand order checked by reading, not run on hardware
//
#include <hip/hip_runtime.h>

// ---------------- problem constants (YOLOv4 head, stride 8) ----------------
#define BATCH   32
#define NA      3
#define FSZ     76
#define NCH     85          // 5 + 80 classes
#define NCLS    80
#define NBOX    50
#define CELLS   (NA * FSZ * FSZ)          // 17328 cells per image
#define TOT     (BATCH * CELLS)           // 554496 total cells
#define INV_STRIDE 0.125f
#define IGNORE_TH  0.7f

__device__ __constant__ float c_anch[9][2] = {
    {12.f,16.f},{19.f,36.f},{40.f,28.f},{36.f,75.f},{76.f,55.f},
    {72.f,146.f},{142.f,110.f},{192.f,243.f},{459.f,401.f}};

// ---------------- helpers ----------------
__device__ __forceinline__ float safe_log(float x) {
    // torch BCELoss: log clamped at -100, arg clamped at tiny
    return fmaxf(logf(fmaxf(x, 1e-43f)), -100.0f);
}

typedef _Float16 v16h __attribute__((ext_vector_type(16)));
typedef float    v8f  __attribute__((ext_vector_type(8)));

// Wave32 sum-reduction via V_WMMA_F32_16X16X32_F16 (D = A x ones + 0).
// fp32 value split into hi/lo f16 halves -> f32 accumulator recovers ~22
// mantissa bits. Requires EXEC == all ones: call with no divergence.
__device__ __forceinline__ float wave_reduce_wmma(float v) {
    _Float16 hi = (_Float16)v;
    _Float16 lo = (_Float16)(v - (float)hi);
    v16h a = {};            // this lane contributes A[m, k0]=hi, A[m, k1]=lo
    a[0] = hi;
    a[1] = lo;
    v16h bv;
#pragma unroll
    for (int i = 0; i < 16; ++i) bv[i] = (_Float16)1.0f;   // all-ones B
    v8f c = {};
    c = __builtin_amdgcn_wmma_f32_16x16x32_f16(
        /*neg_a=*/false, a, /*neg_b=*/false, bv,
        /*c_mod=*/(short)0, c, /*reuse_a=*/false, /*reuse_b=*/false);
    // D[m,n] = row-sum of A; sum my half-column, add partner half-column.
    float s = c[0] + c[1] + c[2] + c[3] + c[4] + c[5] + c[6] + c[7];
    s += __shfl_xor(s, 16, 32);
    return s;               // full wave total, in every lane
}

// Block (256 thr = 8 wave32) reduction; every thread must reach here.
__device__ __forceinline__ float block_reduce(float v, float* sw8) {
    float w = wave_reduce_wmma(v);
    const int lane = threadIdx.x & 31, wv = threadIdx.x >> 5;
    if (lane == 0) sw8[wv] = w;
    __syncthreads();
    float s = 0.0f;
    if (threadIdx.x == 0) {
#pragma unroll
        for (int i = 0; i < 8; ++i) s += sw8[i];
    }
    return s;               // valid in thread 0 only
}

// ---------------- kernel 1: zero the per-cell mask ----------------
__global__ void yolo_zero_mask(float* __restrict__ mask, int n) {
    int i = blockIdx.x * blockDim.x + threadIdx.x;
    if (i < n) mask[i] = 0.0f;
}

// ---------------- kernel 2: anchor match + target scatter ----------------
// One thread per (b, n) truth slot. Writes compact per-cell records.
__global__ void yolo_scatter(const float* __restrict__ labels,
                             float* __restrict__ wsMask,
                             float* __restrict__ wsScale,
                             float* __restrict__ wsFx,
                             float* __restrict__ wsFy,
                             float* __restrict__ wsLw,
                             float* __restrict__ wsLh,
                             float* __restrict__ wsCls) {
    int idx = blockIdx.x * blockDim.x + threadIdx.x;
    if (idx >= BATCH * NBOX) return;
    const int b = idx / NBOX;
    const float* L = labels + (size_t)idx * 5;
    const float cls = L[0], x = L[1], y = L[2], w = L[3], h = L[4];
    const bool valid = (cls + x + y + w + h) > 0.0f;
    const float vf = valid ? 1.0f : 0.0f;

    const float tx = x * INV_STRIDE, ty = y * INV_STRIDE;
    const float tw = w * INV_STRIDE, th = h * INV_STRIDE;
    const float twm = tw * vf, thm = th * vf;
    const float areat = twm * thm;

    // argmax IoU([0,0,w,h], anchors) -- first max wins (matches jnp.argmax)
    float bestIoU = -1.0f;
    int bestK = 0;
#pragma unroll
    for (int k = 0; k < 9; ++k) {
        const float aw = c_anch[k][0] * INV_STRIDE;
        const float ah = c_anch[k][1] * INV_STRIDE;
        const float iw = fminf(twm, aw), ih = fminf(thm, ah);
        const float inter = (iw > 0.0f && ih > 0.0f) ? iw * ih : 0.0f;
        const float iou = inter / (areat + aw * ah - inter);
        if (iou > bestIoU) { bestIoU = iou; bestK = k; }
    }
    const bool bmask = bestK < 3;            // ANCH_MASK = {0,1,2}
    const int ti = (int)tx, tj = (int)ty;
    if (bmask && valid && ti >= 0 && ti < FSZ && tj >= 0 && tj < FSZ) {
        const int bn = bestK;                // == bestK % 3 when < 3
        const size_t cell = (((size_t)b * NA + bn) * FSZ + tj) * FSZ + ti;
        const float maw = c_anch[bn][0] * INV_STRIDE;
        const float mah = c_anch[bn][1] * INV_STRIDE;
        wsMask[cell]  = 1.0f;
        wsScale[cell] = sqrtf(2.0f - tw * th / (float)(FSZ * FSZ));
        wsFx[cell]    = tx - (float)ti;
        wsFy[cell]    = ty - (float)tj;
        wsLw[cell]    = logf(tw / maw + 1e-16f);
        wsLh[cell]    = logf(th / mah + 1e-16f);
        wsCls[cell]   = cls;
    }
}

// ---------------- kernel 3: fused ignore-mask IoU + loss ----------------
// grid = (ceil(CELLS/256), BATCH); block = 256 (8 wave32). Truth boxes in LDS.
__global__ void yolo_loss_main(const float* __restrict__ output,
                               const float* __restrict__ pred,
                               const float* __restrict__ labels,
                               const float* __restrict__ wsMask,
                               const float* __restrict__ wsScale,
                               const float* __restrict__ wsFx,
                               const float* __restrict__ wsFy,
                               const float* __restrict__ wsLw,
                               const float* __restrict__ wsLh,
                               const float* __restrict__ wsCls,
                               float* __restrict__ blockSums) {
    __shared__ float sx1[NBOX], sx2[NBOX], sy1[NBOX], sy2[NBOX], sat[NBOX];
    __shared__ float sw8[8];
    const int tid = threadIdx.x;
    const int b = blockIdx.y;

    if (tid < NBOX) {
        const float* L = labels + ((size_t)b * NBOX + tid) * 5;
        const float cls = L[0], x = L[1], y = L[2], w = L[3], h = L[4];
        const float vf = ((cls + x + y + w + h) > 0.0f) ? 1.0f : 0.0f;
        const float tbx = x * INV_STRIDE * vf, tby = y * INV_STRIDE * vf;
        const float twm = w * INV_STRIDE * vf, thm = h * INV_STRIDE * vf;
        sx1[tid] = tbx - twm * 0.5f;  sx2[tid] = tbx + twm * 0.5f;
        sy1[tid] = tby - thm * 0.5f;  sy2[tid] = tby + thm * 0.5f;
        sat[tid] = twm * thm;
    }
    __syncthreads();

    const int c = blockIdx.x * blockDim.x + tid;
    float partial = 0.0f;
    if (c < CELLS) {
        const size_t cellG = (size_t)b * CELLS + c;
        const float4 pb =
            *reinterpret_cast<const float4*>(pred + cellG * 4);  // 16B aligned
        const float px1 = pb.x - pb.z * 0.5f, px2 = pb.x + pb.z * 0.5f;
        const float py1 = pb.y - pb.w * 0.5f, py2 = pb.y + pb.w * 0.5f;
        const float ap = pb.z * pb.w;

        float best = 0.0f;
        for (int n = 0; n < NBOX; ++n) {      // broadcast LDS reads
            const float lx = fmaxf(px1, sx1[n]), rx = fminf(px2, sx2[n]);
            const float ly = fmaxf(py1, sy1[n]), ry = fminf(py2, sy2[n]);
            const float iw = rx - lx, ih = ry - ly;
            if (iw > 0.0f && ih > 0.0f) {
                const float inter = iw * ih;
                best = fmaxf(best, inter / (ap + sat[n] - inter));
            }
        }
        const float obj0 = (best > IGNORE_TH) ? 0.0f : 1.0f;

        const float m = wsMask[cellG];
        const float objm = (m > 0.0f) ? 1.0f : obj0;   // scatter .set(1.0)
        const float* orow = output + cellG * NCH;
        __builtin_prefetch(orow + 4, 0, 0);            // global_prefetch_b8

        if (objm > 0.0f) {                  // loss_obj; zero when objm == 0
            const float p = orow[4];
            partial -= m * safe_log(p) + (1.0f - m) * safe_log(1.0f - p);
        }
        if (m > 0.0f) {                     // xy / wh / cls only at ~1600 cells
            const float sc = wsScale[cellG], w2 = sc * sc;
            const float fx = wsFx[cellG], fy = wsFy[cellG];
            const float lw = wsLw[cellG], lh = wsLh[cellG];
            const int cls = (int)wsCls[cellG];
            const float p0 = orow[0], p1 = orow[1];
            const float p2 = orow[2], p3 = orow[3];
            partial -= w2 * (fx * safe_log(p0) + (1.0f - fx) * safe_log(1.0f - p0));
            partial -= w2 * (fy * safe_log(p1) + (1.0f - fy) * safe_log(1.0f - p1));
            const float d2 = (p2 - lw) * sc;
            const float d3 = (p3 - lh) * sc;
            partial += 0.5f * (d2 * d2 + d3 * d3);
            for (int k = 0; k < NCLS; ++k) {
                const float p = orow[5 + k];
                const float t = (k == cls) ? 1.0f : 0.0f;
                partial -= t * safe_log(p) + (1.0f - t) * safe_log(1.0f - p);
            }
        }
    }
    // All 256 threads reach here (no early return) -> EXEC all-ones for WMMA.
    const float s = block_reduce(partial, sw8);
    if (tid == 0) blockSums[blockIdx.y * gridDim.x + blockIdx.x] = s;
}

// ---------------- kernel 4: final deterministic reduction ----------------
__global__ void yolo_final_reduce(const float* __restrict__ blockSums, int n,
                                  float* __restrict__ out) {
    __shared__ float sw8[8];
    float v = 0.0f;
    for (int i = threadIdx.x; i < n; i += 256) v += blockSums[i];
    const float s = block_reduce(v, sw8);   // full EXEC here as well
    if (threadIdx.x == 0) out[0] = s;
}

// ---------------- host launcher ----------------
extern "C" void kernel_launch(void* const* d_in, const int* in_sizes, int n_in,
                              void* d_out, int out_size, void* d_ws, size_t ws_size,
                              hipStream_t stream) {
    (void)in_sizes; (void)n_in; (void)out_size; (void)ws_size;
    const float* output = (const float*)d_in[0];
    const float* pred   = (const float*)d_in[1];
    const float* labels = (const float*)d_in[2];
    float* out = (float*)d_out;
    float* ws  = (float*)d_ws;

    // workspace layout (floats): 7 per-cell record arrays + block sums
    float* wsMask  = ws + 0 * (size_t)TOT;
    float* wsScale = ws + 1 * (size_t)TOT;
    float* wsFx    = ws + 2 * (size_t)TOT;
    float* wsFy    = ws + 3 * (size_t)TOT;
    float* wsLw    = ws + 4 * (size_t)TOT;
    float* wsLh    = ws + 5 * (size_t)TOT;
    float* wsCls   = ws + 6 * (size_t)TOT;
    float* wsBlk   = ws + 7 * (size_t)TOT;

    const int gx = (CELLS + 255) / 256;       // 68
    const int nBlk = gx * BATCH;              // 2176

    yolo_zero_mask<<<(TOT + 255) / 256, 256, 0, stream>>>(wsMask, TOT);
    yolo_scatter<<<(BATCH * NBOX + 255) / 256, 256, 0, stream>>>(
        labels, wsMask, wsScale, wsFx, wsFy, wsLw, wsLh, wsCls);
    yolo_loss_main<<<dim3(gx, BATCH), 256, 0, stream>>>(
        output, pred, labels, wsMask, wsScale, wsFx, wsFy, wsLw, wsLh, wsCls,
        wsBlk);
    yolo_final_reduce<<<1, 256, 0, stream>>>(wsBlk, nBlk, out);
}
